// BitLinear_40836549051054
// MI455X (gfx1250) — compile-verified, matrix operand order checked by reading, not run on hardware
//
#include <hip/hip_runtime.h>
#include <hip/hip_bf16.h>

// BitLinear forward for MI455X (gfx1250):
//   RMSNorm -> per-token int8 absmax quant -> int8 x ternary-int8 GEMM via
//   V_WMMA_I32_16X16X64_IU8 with GLOBAL_LOAD_ASYNC_TO_LDS_B128 double-buffered
//   staging -> dequant (1/(scale_x*scale_w)) + bias.

typedef __attribute__((ext_vector_type(8))) int   v8i;
typedef __attribute__((ext_vector_type(4))) float f4;

static constexpr int Mdim = 8 * 2048;  // tokens
static constexpr int Kdim = 4096;      // DIN
static constexpr int Ndim = 4096;      // DOUT

static constexpr int BM = 128;
static constexpr int BN = 128;
static constexpr int BK = 64;          // matches WMMA K for iu8 16x16x64
static constexpr int KT = Kdim / BK;   // 64 k-steps

// ---------------------------------------------------------------------------
// CDNA5 async global->LDS (ASYNCcnt path). vdst = wave-relative LDS byte
// offset (low 32 bits of a generic LDS pointer), vaddr = 64-bit global addr.
// ---------------------------------------------------------------------------
__device__ __forceinline__ void async_copy_b128(unsigned lds_off, const void* gaddr) {
  asm volatile("global_load_async_to_lds_b128 %0, %1, off"
               :: "v"(lds_off), "v"(gaddr) : "memory");
}
__device__ __forceinline__ void wait_async_le4() {
  asm volatile("s_wait_asynccnt 0x4" ::: "memory");
}
__device__ __forceinline__ void wait_async_0() {
  asm volatile("s_wait_asynccnt 0x0" ::: "memory");
}

// ---------------------------------------------------------------------------
// Block-wide reduction helper (wave32; 256 threads = 8 waves)
// ---------------------------------------------------------------------------
__device__ __forceinline__ float blk_reduce(float v, float* red, int t, bool ismax) {
#pragma unroll
  for (int o = 16; o > 0; o >>= 1) {
    float other = __shfl_xor(v, o, 32);
    v = ismax ? fmaxf(v, other) : (v + other);
  }
  if ((t & 31) == 0) red[t >> 5] = v;
  __syncthreads();
  float r = red[0];
#pragma unroll
  for (int w = 1; w < 8; ++w) r = ismax ? fmaxf(r, red[w]) : (r + red[w]);
  __syncthreads();
  return r;
}

// ---------------------------------------------------------------------------
// Kernel 1: fused RMSNorm + absmax int8 quantization (one block per token)
// x is streamed once -> nontemporal loads.
// ---------------------------------------------------------------------------
__global__ __launch_bounds__(256) void quant_kernel(
    const float* __restrict__ x, const float* __restrict__ gamma,
    const float* __restrict__ scale_w, signed char* __restrict__ xq,
    float* __restrict__ rescale) {
  __shared__ float red[8];
  const int t = threadIdx.x;
  const int token = blockIdx.x;

  const f4* xr = (const f4*)(x + (size_t)token * Kdim);
  const f4* gr = (const f4*)gamma;

  f4 xv[4], gv[4];
  float ss = 0.f;
#pragma unroll
  for (int i = 0; i < 4; ++i) {
    const int c = t + i * 256;           // float4 chunk index, coalesced
    xv[i] = __builtin_nontemporal_load(xr + c);
    gv[i] = gr[c];
    ss += xv[i].x * xv[i].x + xv[i].y * xv[i].y +
          xv[i].z * xv[i].z + xv[i].w * xv[i].w;
  }
  ss = blk_reduce(ss, red, t, false);
  const float rstd = rsqrtf(ss * (1.0f / Kdim) + 1e-5f);

  f4 xn[4];
  float amax = 0.f;
#pragma unroll
  for (int i = 0; i < 4; ++i) {
    xn[i] = xv[i] * rstd * gv[i];
    amax = fmaxf(amax, fabsf(xn[i].x));
    amax = fmaxf(amax, fabsf(xn[i].y));
    amax = fmaxf(amax, fabsf(xn[i].z));
    amax = fmaxf(amax, fabsf(xn[i].w));
  }
  amax = blk_reduce(amax, red, t, true);
  const float sx = 127.0f / fmaxf(amax, 1e-5f);

  int* xqi = (int*)(xq + (size_t)token * Kdim);
#pragma unroll
  for (int i = 0; i < 4; ++i) {
    const int c = t + i * 256;
    const int b0 = (int)fminf(fmaxf(rintf(xn[i].x * sx), -128.f), 127.f);
    const int b1 = (int)fminf(fmaxf(rintf(xn[i].y * sx), -128.f), 127.f);
    const int b2 = (int)fminf(fmaxf(rintf(xn[i].z * sx), -128.f), 127.f);
    const int b3 = (int)fminf(fmaxf(rintf(xn[i].w * sx), -128.f), 127.f);
    xqi[c] = (b0 & 0xFF) | ((b1 & 0xFF) << 8) | ((b2 & 0xFF) << 16) | (b3 << 24);
  }
  if (t == 0) rescale[token] = 1.0f / (sx * scale_w[0]);
}

// ---------------------------------------------------------------------------
// Kernel 2: ternary f32 weight -> packed int8 (one packed dword per thread)
// f32 weight is read once -> nontemporal loads. wq stays cached (L2 resident).
// ---------------------------------------------------------------------------
__global__ __launch_bounds__(256) void wconv_kernel(const float* __restrict__ w,
                                                    int* __restrict__ wq) {
  const size_t idx = (size_t)blockIdx.x * 256 + threadIdx.x;  // dword index
  const f4 v = __builtin_nontemporal_load((const f4*)w + idx);
  const int b0 = (int)v.x, b1 = (int)v.y, b2 = (int)v.z, b3 = (int)v.w;
  wq[idx] = (b0 & 0xFF) | ((b1 & 0xFF) << 8) | ((b2 & 0xFF) << 16) | (b3 << 24);
}

// ---------------------------------------------------------------------------
// Fragment load: ISA 8-bit 16x64 layout.
// lane = row (col for B) % 16; half = lane/16.
// VGPR pairs (2j,2j+1) hold 8 consecutive K bytes at half*8 + 16*j.
// ---------------------------------------------------------------------------
__device__ __forceinline__ v8i load_frag(const unsigned char* base) {
  const int2 q0 = *(const int2*)(base + 0);
  const int2 q1 = *(const int2*)(base + 16);
  const int2 q2 = *(const int2*)(base + 32);
  const int2 q3 = *(const int2*)(base + 48);
  v8i f;
  f[0] = q0.x; f[1] = q0.y;
  f[2] = q1.x; f[3] = q1.y;
  f[4] = q2.x; f[5] = q2.y;
  f[6] = q3.x; f[7] = q3.y;
  return f;
}

// ---------------------------------------------------------------------------
// Kernel 3: int8 GEMM  out[m,n] = (sum_k xq[m,k]*wq[n,k]) * rescale[m] + bias[n]
// Block tile 128x128, BK=64, double-buffered LDS via async global->LDS.
// 8 waves: 4(M) x 2(N); each wave = 32x64 slab = 2x4 WMMA tiles.
// ---------------------------------------------------------------------------
__global__ __launch_bounds__(256) void gemm_kernel(
    const signed char* __restrict__ Xq, const signed char* __restrict__ Wq,
    const float* __restrict__ rescale, const float* __restrict__ bias,
    float* __restrict__ out) {
  __shared__ __align__(16) unsigned char sA[2][BM * BK];   // 2 x 8 KB
  __shared__ __align__(16) unsigned char sB[2][BN * BK];   // 2 x 8 KB

  const int t = threadIdx.x;
  const int lane = t & 31;
  const int wave = t >> 5;
  const int half = lane >> 4;
  const int l16 = lane & 15;

  const int waveM = wave & 3;   // 0..3 -> 32-row slice of the block tile
  const int waveN = wave >> 2;  // 0..1 -> 64-col slice

  const int mBase = blockIdx.y * BM;
  const int nBase = blockIdx.x * BN;

  // Per-thread staging assignment: 512 16B chunks per tile, 2 per thread.
  const int q0 = t,        r0 = q0 >> 2, c0 = (q0 & 3) << 4;
  const int q1 = t + 256,  r1 = q1 >> 2, c1 = (q1 & 3) << 4;
  const unsigned offA0 = (unsigned)(r0 * BK + c0);
  const unsigned offA1 = (unsigned)(r1 * BK + c1);
  const signed char* gA0 = Xq + (size_t)(mBase + r0) * Kdim + c0;
  const signed char* gA1 = Xq + (size_t)(mBase + r1) * Kdim + c1;
  const signed char* gB0 = Wq + (size_t)(nBase + r0) * Kdim + c0;
  const signed char* gB1 = Wq + (size_t)(nBase + r1) * Kdim + c1;

  const unsigned sAoff = (unsigned)(size_t)&sA[0][0];  // LDS-relative base
  const unsigned sBoff = (unsigned)(size_t)&sB[0][0];

  auto issue_stage = [&](int buf, int k0) {
    const unsigned ab = sAoff + (unsigned)buf * (BM * BK);
    const unsigned bb = sBoff + (unsigned)buf * (BN * BK);
    async_copy_b128(ab + offA0, gA0 + k0);
    async_copy_b128(ab + offA1, gA1 + k0);
    async_copy_b128(bb + offA0, gB0 + k0);
    async_copy_b128(bb + offA1, gB1 + k0);
  };

  v8i acc[2][4];
  {
    const v8i zero = {};
#pragma unroll
    for (int i = 0; i < 2; ++i)
#pragma unroll
      for (int j = 0; j < 4; ++j) acc[i][j] = zero;
  }

  issue_stage(0, 0);  // prologue

  for (int kb = 0; kb < KT; ++kb) {
    const int cur = kb & 1;
    if (kb + 1 < KT) {
      issue_stage(cur ^ 1, (kb + 1) * BK);  // next stage in flight
      wait_async_le4();                     // in-order: first 4 (current) done
    } else {
      wait_async_0();
    }
    __syncthreads();                        // current buffer visible to all waves

    v8i af[2], bf[4];
#pragma unroll
    for (int i = 0; i < 2; ++i)
      af[i] = load_frag(&sA[cur][(waveM * 32 + i * 16 + l16) * BK + half * 8]);
#pragma unroll
    for (int j = 0; j < 4; ++j)
      bf[j] = load_frag(&sB[cur][(waveN * 64 + j * 16 + l16) * BK + half * 8]);

#pragma unroll
    for (int i = 0; i < 2; ++i)
#pragma unroll
      for (int j = 0; j < 4; ++j)
        acc[i][j] = __builtin_amdgcn_wmma_i32_16x16x64_iu8(
            true, af[i], true, bf[j], acc[i][j], false, false);

    __syncthreads();                        // done reading before buf is rewritten
  }

  // Epilogue: C/D layout -> lane holds column l16, rows half*8 + r (r = vgpr idx)
  float rs[2][8];
#pragma unroll
  for (int i = 0; i < 2; ++i)
#pragma unroll
    for (int r = 0; r < 8; ++r)
      rs[i][r] = rescale[mBase + waveM * 32 + i * 16 + half * 8 + r];

#pragma unroll
  for (int j = 0; j < 4; ++j) {
    const int gn = nBase + waveN * 64 + j * 16 + l16;
    const float bn = bias[gn];
#pragma unroll
    for (int i = 0; i < 2; ++i) {
      const int gm0 = mBase + waveM * 32 + i * 16 + half * 8;
#pragma unroll
      for (int r = 0; r < 8; ++r)
        __builtin_nontemporal_store((float)acc[i][j][r] * rs[i][r] + bn,
                                    &out[(size_t)(gm0 + r) * Ndim + gn]);
    }
  }
}

// ---------------------------------------------------------------------------
// Launch
// ---------------------------------------------------------------------------
extern "C" void kernel_launch(void* const* d_in, const int* in_sizes, int n_in,
                              void* d_out, int out_size, void* d_ws, size_t ws_size,
                              hipStream_t stream) {
  const float* x       = (const float*)d_in[0];
  const float* w       = (const float*)d_in[1];
  const float* scale_w = (const float*)d_in[2];
  const float* gamma   = (const float*)d_in[3];
  const float* bias    = (const float*)d_in[4];
  float* out           = (float*)d_out;

  char* ws = (char*)d_ws;
  signed char* xq = (signed char*)ws;                                   // M*K int8
  signed char* wq = (signed char*)(ws + (size_t)Mdim * Kdim);           // N*K int8
  float* rescale  = (float*)(ws + (size_t)Mdim * Kdim + (size_t)Ndim * Kdim);

  // 1) RMSNorm + quantize activations, one block per token
  quant_kernel<<<Mdim, 256, 0, stream>>>(x, gamma, scale_w, xq, rescale);

  // 2) ternary weight -> int8 (N*K/4 dwords, 256 threads/block)
  wconv_kernel<<<(Ndim * Kdim / 4) / 256, 256, 0, stream>>>(w, (int*)wq);

  // 3) int8 WMMA GEMM + dequant + bias
  dim3 grid(Ndim / BN, Mdim / BM);
  gemm_kernel<<<grid, 256, 0, stream>>>(xq, wq, rescale, bias, out);
}
